// ModelTrunk_19653770346864
// MI455X (gfx1250) — matrix operand order repaired and checked by reading, not compiled, not run on hardware
//
#include <hip/hip_runtime.h>
#include <hip/hip_bf16.h>
#include <math.h>

// ---------------------------------------------------------------------------
// CDNA5 (gfx1250) wave32 WMMA bf16 transformer block.
// All GEMMs use v_wmma_f32_16x16x32_bf16.
// ---------------------------------------------------------------------------

typedef __attribute__((ext_vector_type(16))) __bf16 v16bf;
typedef __attribute__((ext_vector_type(8)))  float  v8f;

#define DEV __device__ __forceinline__

constexpr int Bsz  = 4;
constexpr int Tseq = 2048;
constexpr int Fdim = 128;
constexpr int Dm   = 384;      // d_model
constexpr int Hh   = 8;        // heads
constexpr int HSd  = 128;      // head size
constexpr int FFd  = 512;      // ffn dim
constexpr int Mrows = Bsz * Tseq;   // 8192

union Frag16 { v16bf v; uint4 q[2]; __bf16 e[16]; };

DEV v8f zero8() {
  v8f z;
  #pragma unroll
  for (int i = 0; i < 8; ++i) z[i] = 0.f;
  return z;
}

// A fragment: 16x32 (MxK) bf16, row-major A[m][k], lda in elements.
// lane<16: e[0..7]=K(k0..k0+7), e[8..15]=K(k0+16..k0+23)
// lane>=16: shifted by 8 in K.  (ISA 7.12.2 16-bit A layout)
DEV v16bf load_a(const __bf16* __restrict__ A, int lda, int m0, int k0, int lane) {
  const int m = m0 + (lane & 15);
  const int half = lane >> 4;
  const __bf16* p = A + (size_t)m * lda + k0 + half * 8;
  Frag16 f;
  f.q[0] = *(const uint4*)p;
  f.q[1] = *(const uint4*)(p + 16);
  return f.v;
}

// B fragment: 32x16 (KxN) bf16 from *transposed* storage Bt[n][k], ldb elems.
// lane = N (mod 16); e[0..7] -> K = k0 + 8*(lane>>4) + j, e[8..15] -> +16 (same as A).
DEV v16bf load_b(const __bf16* __restrict__ Bt, int ldb, int n0, int k0, int lane) {
  const int n = n0 + (lane & 15);
  const int half = lane >> 4;
  const __bf16* p = Bt + (size_t)n * ldb + k0 + half * 8;
  Frag16 f;
  f.q[0] = *(const uint4*)p;
  f.q[1] = *(const uint4*)(p + 16);
  return f.v;
}

// Wave-level GEMM: C(16 x 16*NF) += A(16xK) * B(KxN), K multiple of 32.
template <int NF>
DEV void gemm_acc(const __bf16* __restrict__ A, int lda,
                  const __bf16* __restrict__ Bt, int ldb,
                  int m0, int n0, int K, int lane, v8f* acc) {
  for (int k0 = 0; k0 < K; k0 += 32) {
    v16bf a = load_a(A, lda, m0, k0, lane);
    #pragma unroll
    for (int i = 0; i < NF; ++i) {
      v16bf b = load_b(Bt, ldb, n0 + 16 * i, k0, lane);
      acc[i] = __builtin_amdgcn_wmma_f32_16x16x32_bf16(
          false, a, false, b, (short)0, acc[i], false, false);
    }
  }
}

// ---------------------------------------------------------------------------
// Weight pack: in [heads][K][N] fp32  ->  out [heads][N][K] bf16
// ---------------------------------------------------------------------------
__global__ void transpose_bf16_kernel(const float* __restrict__ in,
                                      __bf16* __restrict__ out, int K, int N) {
  const size_t hoff = (size_t)blockIdx.y * K * N;
  const int total = K * N;
  for (int o = blockIdx.x * blockDim.x + threadIdx.x; o < total;
       o += gridDim.x * blockDim.x) {
    const int kk = o % K;
    const int n = o / K;
    out[hoff + o] = (__bf16)in[hoff + (size_t)kk * N + n];
  }
}

// ---------------------------------------------------------------------------
// Time2Vec: h[row] = [ x(128) | interleave(wb*x+bb, sin(x.wa + ba)) (256) ]
// one block (128 thr) per row; fp32 + bf16 copies.
// ---------------------------------------------------------------------------
__global__ __launch_bounds__(128)
void t2v_kernel(const float* __restrict__ x, const float* __restrict__ wb,
                const float* __restrict__ bb, const float* __restrict__ wa,
                const float* __restrict__ ba, float* __restrict__ hf,
                __bf16* __restrict__ hb) {
  const int row = blockIdx.x;
  const int f = threadIdx.x;
  const int lane = f & 31, w = f >> 5;
  const float xv = x[(size_t)row * Fdim + f];
  float part = xv * wa[f];
  #pragma unroll
  for (int off = 16; off > 0; off >>= 1) part += __shfl_xor(part, off, 32);
  __shared__ float red[4];
  if (lane == 0) red[w] = part;
  __syncthreads();
  const float dot = red[0] + red[1] + red[2] + red[3];
  const float bias = wb[f] * xv + bb[f];
  const float sv = __sinf(dot + ba[f]);
  const size_t base = (size_t)row * Dm;
  hf[base + f] = xv;                    hb[base + f] = (__bf16)xv;
  hf[base + Fdim + 2 * f] = bias;       hb[base + Fdim + 2 * f] = (__bf16)bias;
  hf[base + Fdim + 2 * f + 1] = sv;     hb[base + Fdim + 2 * f + 1] = (__bf16)sv;
}

// ---------------------------------------------------------------------------
// Fused QKV GEMM: [8192 x 384] x [384 x 3072] (per-head q|k|v, 128 cols each)
// block 256 thr = 8 waves (2M x 4N) -> 32 x 256 tile. grid (256, 12).
// q scaled by 1/sqrt(HS); v stored transposed [b,h][hs][t].
// ---------------------------------------------------------------------------
__global__ __launch_bounds__(256)
void qkv_kernel(const __bf16* __restrict__ hbf, const __bf16* __restrict__ Wq_t,
                const __bf16* __restrict__ Wk_t, const __bf16* __restrict__ Wv_t,
                __bf16* __restrict__ q, __bf16* __restrict__ k,
                __bf16* __restrict__ vt) {
  const int lane = threadIdx.x & 31;
  const int w = threadIdx.x >> 5;
  const int half = lane >> 4, c = lane & 15;
  const int m0 = blockIdx.x * 32 + (w >> 2) * 16;
  const int ng = blockIdx.y * 256 + (w & 3) * 64;   // global column
  const int g = ng / 128;                            // 128-col group
  const int h = g / 3, which = g % 3;                // head, q/k/v
  const int ns = ng % 128;                           // col within head
  const __bf16* Bt =
      (which == 0 ? Wq_t : which == 1 ? Wk_t : Wv_t) + (size_t)h * HSd * Dm;

  v8f acc[4];
  #pragma unroll
  for (int i = 0; i < 4; ++i) acc[i] = zero8();
  gemm_acc<4>(hbf, Dm, Bt, Dm, m0, ns, Dm, lane, acc);

  constexpr float qscale = 0.08838834764831845f;   // 1/sqrt(128)
  #pragma unroll
  for (int r = 0; r < 8; ++r) {
    const int m = m0 + r + 8 * half;
    const int bidx = m / Tseq, t = m % Tseq;
    #pragma unroll
    for (int i = 0; i < 4; ++i) {
      const int hs = ns + i * 16 + c;
      const float v = acc[i][r];
      if (which == 0)
        q[((size_t)(bidx * Hh + h) * Tseq + t) * HSd + hs] = (__bf16)(v * qscale);
      else if (which == 1)
        k[((size_t)(bidx * Hh + h) * Tseq + t) * HSd + hs] = (__bf16)v;
      else
        vt[((size_t)(bidx * Hh + h) * HSd + hs) * Tseq + t] = (__bf16)v;
    }
  }
}

// ---------------------------------------------------------------------------
// Flash attention. grid (T/128, B*H); 8 waves, each owns 16 queries x HS=128.
// Streams 64-key tiles: S = QK^T (wmma), streaming softmax (shfl_xor row
// reductions over the 16-lane halves), P -> LDS (C-layout -> A-layout),
// O += P V (wmma).
// ---------------------------------------------------------------------------
__global__ __launch_bounds__(256)
void attn_kernel(const __bf16* __restrict__ q, const __bf16* __restrict__ k,
                 const __bf16* __restrict__ vt, __bf16* __restrict__ mo) {
  __shared__ __align__(16) __bf16 plds[8][16][80];   // wave-private P tiles
  const int lane = threadIdx.x & 31;
  const int w = threadIdx.x >> 5;
  const int half = lane >> 4, c = lane & 15;
  const int m0 = blockIdx.x * 128 + w * 16;          // query tile (within T)
  const int bh = blockIdx.y;
  const int b = bh / Hh, h = bh % Hh;
  const __bf16* qb = q + (size_t)bh * Tseq * HSd;
  const __bf16* kb = k + (size_t)bh * Tseq * HSd;
  const __bf16* vb = vt + (size_t)bh * HSd * Tseq;

  v8f o[8];
  #pragma unroll
  for (int i = 0; i < 8; ++i) o[i] = zero8();
  float mrow[8], lrow[8];
  #pragma unroll
  for (int r = 0; r < 8; ++r) { mrow[r] = -3.0e38f; lrow[r] = 0.f; }

  for (int kt = 0; kt < Tseq; kt += 64) {
    // ---- S = Q K^T : 16 x 64 tile
    v8f s[4];
    #pragma unroll
    for (int i = 0; i < 4; ++i) s[i] = zero8();
    #pragma unroll
    for (int k0 = 0; k0 < HSd; k0 += 32) {
      v16bf a = load_a(qb, HSd, m0, k0, lane);
      #pragma unroll
      for (int i = 0; i < 4; ++i) {
        v16bf bf = load_b(kb, HSd, kt + 16 * i, k0, lane);
        s[i] = __builtin_amdgcn_wmma_f32_16x16x32_bf16(
            false, a, false, bf, (short)0, s[i], false, false);
      }
    }
    // ---- streaming softmax (row = r + 8*half, cols across 16 lanes x 4 frags)
    #pragma unroll
    for (int r = 0; r < 8; ++r) {
      float mx = fmaxf(fmaxf(s[0][r], s[1][r]), fmaxf(s[2][r], s[3][r]));
      mx = fmaxf(mx, __shfl_xor(mx, 1, 32));
      mx = fmaxf(mx, __shfl_xor(mx, 2, 32));
      mx = fmaxf(mx, __shfl_xor(mx, 4, 32));
      mx = fmaxf(mx, __shfl_xor(mx, 8, 32));
      const float mnew = fmaxf(mrow[r], mx);
      const float corr = __expf(mrow[r] - mnew);
      mrow[r] = mnew;
      float rs = 0.f;
      #pragma unroll
      for (int i = 0; i < 4; ++i) {
        const float p = __expf(s[i][r] - mnew);
        s[i][r] = p;
        rs += p;
      }
      rs += __shfl_xor(rs, 1, 32);
      rs += __shfl_xor(rs, 2, 32);
      rs += __shfl_xor(rs, 4, 32);
      rs += __shfl_xor(rs, 8, 32);
      lrow[r] = lrow[r] * corr + rs;
      #pragma unroll
      for (int i = 0; i < 8; ++i) o[i][r] *= corr;
    }
    // ---- stage P (C-layout) into LDS, re-read in A-layout
    #pragma unroll
    for (int i = 0; i < 4; ++i)
      #pragma unroll
      for (int r = 0; r < 8; ++r)
        plds[w][r + 8 * half][i * 16 + c] = (__bf16)s[i][r];
    // ---- O += P V
    #pragma unroll
    for (int k0 = 0; k0 < 64; k0 += 32) {
      Frag16 fa;
      const __bf16* pp = &plds[w][c][k0 + half * 8];
      fa.q[0] = *(const uint4*)pp;
      fa.q[1] = *(const uint4*)(pp + 16);
      #pragma unroll
      for (int i = 0; i < 8; ++i) {
        v16bf bf = load_b(vb, Tseq, i * 16, kt + k0, lane);
        o[i] = __builtin_amdgcn_wmma_f32_16x16x32_bf16(
            false, fa.v, false, bf, (short)0, o[i], false, false);
      }
    }
  }
  // ---- epilogue: mo[b][t][h][hs] = O / l
  #pragma unroll
  for (int r = 0; r < 8; ++r) {
    const int t = m0 + r + 8 * half;
    const float inv = 1.f / lrow[r];
    const size_t base = ((size_t)(b * Tseq + t) * Hh + h) * HSd;
    #pragma unroll
    for (int i = 0; i < 8; ++i)
      mo[base + i * 16 + c] = (__bf16)(o[i][r] * inv);
  }
}

// ---------------------------------------------------------------------------
// Output projection: [8192 x 1024] x [1024 x 384] + bp -> abuf (fp32)
// 8 waves (4M x 2N) -> 64 x 128 tile. grid (128, 3).
// ---------------------------------------------------------------------------
__global__ __launch_bounds__(256)
void proj_kernel(const __bf16* __restrict__ mo, const __bf16* __restrict__ Wp_t,
                 const float* __restrict__ bp, float* __restrict__ abuf) {
  const int lane = threadIdx.x & 31;
  const int w = threadIdx.x >> 5;
  const int half = lane >> 4, c = lane & 15;
  const int m0 = blockIdx.x * 64 + (w >> 1) * 16;
  const int n0 = blockIdx.y * 128 + (w & 1) * 64;
  v8f acc[4];
  #pragma unroll
  for (int i = 0; i < 4; ++i) acc[i] = zero8();
  gemm_acc<4>(mo, Hh * HSd, Wp_t, Hh * HSd, m0, n0, Hh * HSd, lane, acc);
  #pragma unroll
  for (int r = 0; r < 8; ++r) {
    const int m = m0 + r + 8 * half;
    #pragma unroll
    for (int i = 0; i < 4; ++i) {
      const int n = n0 + i * 16 + c;
      abuf[(size_t)m * Dm + n] = acc[i][r] + bp[n];
    }
  }
}

// ---------------------------------------------------------------------------
// FFN1: [8192 x 384] x [384 x 512] + b1, ReLU -> bf16.  32x256 tile, grid (256,2)
// ---------------------------------------------------------------------------
__global__ __launch_bounds__(256)
void ffn1_kernel(const __bf16* __restrict__ h1b, const __bf16* __restrict__ W1_t,
                 const float* __restrict__ b1, __bf16* __restrict__ fb) {
  const int lane = threadIdx.x & 31;
  const int w = threadIdx.x >> 5;
  const int half = lane >> 4, c = lane & 15;
  const int m0 = blockIdx.x * 32 + (w >> 2) * 16;
  const int n0 = blockIdx.y * 256 + (w & 3) * 64;
  v8f acc[4];
  #pragma unroll
  for (int i = 0; i < 4; ++i) acc[i] = zero8();
  gemm_acc<4>(h1b, Dm, W1_t, Dm, m0, n0, Dm, lane, acc);
  #pragma unroll
  for (int r = 0; r < 8; ++r) {
    const int m = m0 + r + 8 * half;
    #pragma unroll
    for (int i = 0; i < 4; ++i) {
      const int n = n0 + i * 16 + c;
      fb[(size_t)m * FFd + n] = (__bf16)fmaxf(acc[i][r] + b1[n], 0.f);
    }
  }
}

// ---------------------------------------------------------------------------
// FFN2: [8192 x 512] x [512 x 384] + b2 -> fp32.  64x128 tile, grid (128,3)
// ---------------------------------------------------------------------------
__global__ __launch_bounds__(256)
void ffn2_kernel(const __bf16* __restrict__ fb, const __bf16* __restrict__ W2_t,
                 const float* __restrict__ b2, float* __restrict__ f2) {
  const int lane = threadIdx.x & 31;
  const int w = threadIdx.x >> 5;
  const int half = lane >> 4, c = lane & 15;
  const int m0 = blockIdx.x * 64 + (w >> 1) * 16;
  const int n0 = blockIdx.y * 128 + (w & 1) * 64;
  v8f acc[4];
  #pragma unroll
  for (int i = 0; i < 4; ++i) acc[i] = zero8();
  gemm_acc<4>(fb, FFd, W2_t, FFd, m0, n0, FFd, lane, acc);
  #pragma unroll
  for (int r = 0; r < 8; ++r) {
    const int m = m0 + r + 8 * half;
    #pragma unroll
    for (int i = 0; i < 4; ++i) {
      const int n = n0 + i * 16 + c;
      f2[(size_t)m * Dm + n] = acc[i][r] + b2[n];
    }
  }
}

// ---------------------------------------------------------------------------
// y = LayerNorm(a + b) * g + beta ; writes fp32 (+optional bf16 copy).
// one block (128 thr) per row of 384; each thread owns 3 elements.
// ---------------------------------------------------------------------------
__global__ __launch_bounds__(128)
void add_ln_kernel(const float* __restrict__ a, const float* __restrict__ bsrc,
                   const float* __restrict__ g, const float* __restrict__ beta,
                   float* __restrict__ yf, __bf16* __restrict__ yb) {
  const int row = blockIdx.x;
  const int d = threadIdx.x;
  const int lane = d & 31, w = d >> 5;
  const size_t base = (size_t)row * Dm;
  const float s0 = a[base + d] + bsrc[base + d];
  const float s1 = a[base + d + 128] + bsrc[base + d + 128];
  const float s2 = a[base + d + 256] + bsrc[base + d + 256];
  float sum = s0 + s1 + s2;
  float sq = s0 * s0 + s1 * s1 + s2 * s2;
  #pragma unroll
  for (int off = 16; off > 0; off >>= 1) {
    sum += __shfl_xor(sum, off, 32);
    sq += __shfl_xor(sq, off, 32);
  }
  __shared__ float rs[4], rq[4];
  if (lane == 0) { rs[w] = sum; rq[w] = sq; }
  __syncthreads();
  sum = rs[0] + rs[1] + rs[2] + rs[3];
  sq = rq[0] + rq[1] + rq[2] + rq[3];
  const float mean = sum * (1.f / Dm);
  const float var = sq * (1.f / Dm) - mean * mean;
  const float rstd = rsqrtf(var + 1e-6f);
  const float y0 = g[d] * (s0 - mean) * rstd + beta[d];
  const float y1 = g[d + 128] * (s1 - mean) * rstd + beta[d + 128];
  const float y2 = g[d + 256] * (s2 - mean) * rstd + beta[d + 256];
  yf[base + d] = y0;
  yf[base + d + 128] = y1;
  yf[base + d + 256] = y2;
  if (yb) {
    yb[base + d] = (__bf16)y0;
    yb[base + d + 128] = (__bf16)y1;
    yb[base + d + 256] = (__bf16)y2;
  }
}

// ---------------------------------------------------------------------------
extern "C" void kernel_launch(void* const* d_in, const int* in_sizes, int n_in,
                              void* d_out, int out_size, void* d_ws,
                              size_t ws_size, hipStream_t stream) {
  (void)in_sizes; (void)n_in; (void)out_size; (void)ws_size;
  const float* x    = (const float*)d_in[0];
  const float* wb   = (const float*)d_in[1];
  const float* bb   = (const float*)d_in[2];
  const float* wa   = (const float*)d_in[3];
  const float* ba   = (const float*)d_in[4];
  const float* Wq   = (const float*)d_in[5];
  const float* Wk   = (const float*)d_in[6];
  const float* Wv   = (const float*)d_in[7];
  const float* Wp   = (const float*)d_in[8];
  const float* bp   = (const float*)d_in[9];
  const float* ln1g = (const float*)d_in[10];
  const float* ln1b = (const float*)d_in[11];
  const float* W1   = (const float*)d_in[12];
  const float* b1   = (const float*)d_in[13];
  const float* W2   = (const float*)d_in[14];
  const float* b2   = (const float*)d_in[15];
  const float* ln2g = (const float*)d_in[16];
  const float* ln2b = (const float*)d_in[17];
  float* out = (float*)d_out;

  // ---- workspace layout
  char* ws = (char*)d_ws;
  size_t off = 0;
  auto take = [&](size_t bytes) {
    size_t o = off;
    off = (off + bytes + 255) & ~(size_t)255;
    return o;
  };
  float*  hf   = (float*)(ws + take((size_t)Mrows * Dm * 4));
  __bf16* hb   = (__bf16*)(ws + take((size_t)Mrows * Dm * 2));
  __bf16* qbuf = (__bf16*)(ws + take((size_t)Bsz * Hh * Tseq * HSd * 2));
  __bf16* kbuf = (__bf16*)(ws + take((size_t)Bsz * Hh * Tseq * HSd * 2));
  __bf16* vtb  = (__bf16*)(ws + take((size_t)Bsz * Hh * HSd * Tseq * 2));
  __bf16* mob  = (__bf16*)(ws + take((size_t)Mrows * Hh * HSd * 2));
  float*  abuf = (float*)(ws + take((size_t)Mrows * Dm * 4));
  float*  h1f  = (float*)(ws + take((size_t)Mrows * Dm * 4));
  __bf16* h1b  = (__bf16*)(ws + take((size_t)Mrows * Dm * 2));
  __bf16* fbuf = (__bf16*)(ws + take((size_t)Mrows * FFd * 2));
  float*  f2   = (float*)(ws + take((size_t)Mrows * Dm * 4));
  __bf16* wqt  = (__bf16*)(ws + take((size_t)Hh * HSd * Dm * 2));
  __bf16* wkt  = (__bf16*)(ws + take((size_t)Hh * HSd * Dm * 2));
  __bf16* wvt  = (__bf16*)(ws + take((size_t)Hh * HSd * Dm * 2));
  __bf16* wpt  = (__bf16*)(ws + take((size_t)Dm * Hh * HSd * 2));
  __bf16* w1t  = (__bf16*)(ws + take((size_t)FFd * Dm * 2));
  __bf16* w2t  = (__bf16*)(ws + take((size_t)Dm * FFd * 2));

  // ---- pack weights (fp32 -> bf16, transposed to [N][K])
  transpose_bf16_kernel<<<dim3(192, Hh), 256, 0, stream>>>(Wq, wqt, Dm, HSd);
  transpose_bf16_kernel<<<dim3(192, Hh), 256, 0, stream>>>(Wk, wkt, Dm, HSd);
  transpose_bf16_kernel<<<dim3(192, Hh), 256, 0, stream>>>(Wv, wvt, Dm, HSd);
  transpose_bf16_kernel<<<dim3(1536, 1), 256, 0, stream>>>(Wp, wpt, Hh * HSd, Dm);
  transpose_bf16_kernel<<<dim3(768, 1), 256, 0, stream>>>(W1, w1t, Dm, FFd);
  transpose_bf16_kernel<<<dim3(768, 1), 256, 0, stream>>>(W2, w2t, FFd, Dm);

  // ---- time2vec -> h (fp32 + bf16)
  t2v_kernel<<<dim3(Mrows), 128, 0, stream>>>(x, wb, bb, wa, ba, hf, hb);

  // ---- fused QKV projection
  qkv_kernel<<<dim3(Mrows / 32, 12), 256, 0, stream>>>(hb, wqt, wkt, wvt,
                                                       qbuf, kbuf, vtb);

  // ---- flash attention
  attn_kernel<<<dim3(Tseq / 128, Bsz * Hh), 256, 0, stream>>>(qbuf, kbuf, vtb,
                                                              mob);

  // ---- output projection + LN1
  proj_kernel<<<dim3(Mrows / 64, Dm / 128), 256, 0, stream>>>(mob, wpt, bp,
                                                              abuf);
  add_ln_kernel<<<dim3(Mrows), 128, 0, stream>>>(hf, abuf, ln1g, ln1b, h1f,
                                                 h1b);

  // ---- FFN + LN2 (writes d_out)
  ffn1_kernel<<<dim3(Mrows / 32, FFd / 256), 256, 0, stream>>>(h1b, w1t, b1,
                                                               fbuf);
  ffn2_kernel<<<dim3(Mrows / 64, Dm / 128), 256, 0, stream>>>(fbuf, w2t, b2,
                                                              f2);
  add_ln_kernel<<<dim3(Mrows), 128, 0, stream>>>(h1f, f2, ln2g, ln2b, out,
                                                 nullptr);
}
